// FSAS_34574486732937
// MI455X (gfx1250) — matrix operand order repaired and checked by reading, not compiled, not run on hardware
//
#include <hip/hip_runtime.h>
#include <hip/hip_bf16.h>

// MI455X / gfx1250 implementation.
// Pipeline:
//  1) proj_wmma     : ql/kl/vl = x @ W.T + b via v_wmma_f32_16x16x32_f16,
//                     written directly in head-split layout [B,H,N,32]
//  2) rfft32_rows   : 32-point rfft along head_dim per row -> [plane][f=17][n=4096]
//  3) fft4096_fwd   : in-LDS 4096-pt DIF FFT per (plane,f) column (bit-reversed out)
//  4) ifft4096_mul  : elementwise Qf*Kf then in-LDS DIT inverse FFT (natural out)
//  5) attn_gate     : 32-pt irfft per row + softmax(32) + gate with vl -> [B,N,256]
//  6) residual_ln   : +v residual, LayerNorm(256), one wave32 per token

typedef __attribute__((ext_vector_type(16))) _Float16 v16h;
typedef __attribute__((ext_vector_type(8)))  float    v8f;

#define DIMC   256
#define SEQN   4096
#define NHEAD  8
#define HDIM   32
#define NPLANE 64            // B*H = 8*8
#define NFREQ  17            // HDIM/2 + 1
#define PI_F   3.14159265358979323846f
#define TWO_PI 6.28318530717958647692f

// ---------------------------------------------------------------------------
// Fragment loaders; A and B both use contiguous k order k = 16*(lane/16) + s.
// A 16x32 f16: lanes 0-15 hold M=lane, halves = K0..15;
//              lanes 16-31 hold M=lane-16, halves = K16..31.
__device__ __forceinline__ v16h load_a_frag(const float* __restrict__ arow,
                                            int kc, int khi) {
    const float4* p0 = (const float4*)(arow + kc + khi);
    const float4* p1 = (const float4*)(arow + kc + 8 + khi);
    float4 f0 = p0[0], f1 = p0[1], f2 = p1[0], f3 = p1[1];
    v16h a;
    a[0]=(_Float16)f0.x; a[1]=(_Float16)f0.y; a[2]=(_Float16)f0.z; a[3]=(_Float16)f0.w;
    a[4]=(_Float16)f1.x; a[5]=(_Float16)f1.y; a[6]=(_Float16)f1.z; a[7]=(_Float16)f1.w;
    a[8]=(_Float16)f2.x; a[9]=(_Float16)f2.y; a[10]=(_Float16)f2.z; a[11]=(_Float16)f2.w;
    a[12]=(_Float16)f3.x; a[13]=(_Float16)f3.y; a[14]=(_Float16)f3.z; a[15]=(_Float16)f3.w;
    return a;
}
// B 32x16 f16: lanes 0-15 hold column N=lane, halves = K0..15;
//              lanes 16-31 hold column N=lane-16, halves = K16..31.
// B[k][n] = W[n_global][k_global] (out = x @ W.T), so a lane reads 16
// contiguous floats of one W row.
__device__ __forceinline__ v16h load_b_frag(const float* __restrict__ wrow, int kb) {
    const float4* p = (const float4*)(wrow + kb);
    float4 f0 = p[0], f1 = p[1], f2 = p[2], f3 = p[3];
    v16h b;
    b[0]=(_Float16)f0.x; b[1]=(_Float16)f0.y; b[2]=(_Float16)f0.z; b[3]=(_Float16)f0.w;
    b[4]=(_Float16)f1.x; b[5]=(_Float16)f1.y; b[6]=(_Float16)f1.z; b[7]=(_Float16)f1.w;
    b[8]=(_Float16)f2.x; b[9]=(_Float16)f2.y; b[10]=(_Float16)f2.z; b[11]=(_Float16)f2.w;
    b[12]=(_Float16)f3.x; b[13]=(_Float16)f3.y; b[14]=(_Float16)f3.z; b[15]=(_Float16)f3.w;
    return b;
}

// ---------------------------------------------------------------------------
// Kernel 1: fused QKV projection. grid(1024, 3) x 256 threads (8 wave32).
// Each wave computes a 16x64 output strip (4 WMMA N-tiles, A fragment reused).
__global__ __launch_bounds__(256) void proj_wmma(
    const float* __restrict__ q, const float* __restrict__ k, const float* __restrict__ v,
    const float* __restrict__ Wq, const float* __restrict__ bq,
    const float* __restrict__ Wk, const float* __restrict__ bk,
    const float* __restrict__ Wv, const float* __restrict__ bv,
    float* __restrict__ ql, float* __restrict__ kl, float* __restrict__ vl)
{
    const int proj = blockIdx.y;
    const float *X, *W, *bias;
    float* OUT;
    if (proj == 0)      { X = q; W = Wq; bias = bq; OUT = ql; }
    else if (proj == 1) { X = k; W = Wk; bias = bk; OUT = kl; }
    else                { X = v; W = Wv; bias = bv; OUT = vl; }

    const int wave = threadIdx.x >> 5;
    const int lane = threadIdx.x & 31;
    const int tileIdx = blockIdx.x * 8 + wave;   // 0..8191
    const int mTile   = tileIdx >> 2;            // 0..2047 (16 rows each)
    const int colGrp  = tileIdx & 3;             // 0..3 (64 cols each)

    const int row  = (mTile << 4) + (lane & 15);
    const int khi  = (lane >> 4) << 4;           // A-frag K offset: 0 or 16
    const int kbB  = (lane >> 4) << 4;           // B-frag K offset: 0 or 16
    const int colBase = colGrp << 6;
    const int col0 = colBase + (lane & 15);

    const float* arow = X + (size_t)row * DIMC;
    const float* w0 = W + (size_t)(col0 +  0) * DIMC;
    const float* w1 = W + (size_t)(col0 + 16) * DIMC;
    const float* w2 = W + (size_t)(col0 + 32) * DIMC;
    const float* w3 = W + (size_t)(col0 + 48) * DIMC;

    v8f acc0 = {}, acc1 = {}, acc2 = {}, acc3 = {};

    #pragma unroll
    for (int kc = 0; kc < DIMC; kc += 32) {
        if (kc + 32 < DIMC)
            __builtin_prefetch(arow + kc + 32, 0, 1);   // global_prefetch_b8
        v16h a  = load_a_frag(arow, kc, khi);
        v16h b0 = load_b_frag(w0, kc + kbB);
        v16h b1 = load_b_frag(w1, kc + kbB);
        v16h b2 = load_b_frag(w2, kc + kbB);
        v16h b3 = load_b_frag(w3, kc + kbB);
        acc0 = __builtin_amdgcn_wmma_f32_16x16x32_f16(false, a, false, b0, (short)0, acc0, false, false);
        acc1 = __builtin_amdgcn_wmma_f32_16x16x32_f16(false, a, false, b1, (short)0, acc1, false, false);
        acc2 = __builtin_amdgcn_wmma_f32_16x16x32_f16(false, a, false, b2, (short)0, acc2, false, false);
        acc3 = __builtin_amdgcn_wmma_f32_16x16x32_f16(false, a, false, b3, (short)0, acc3, false, false);
    }

    const float bv0 = bias[col0], bv1 = bias[col0 + 16],
                bv2 = bias[col0 + 32], bv3 = bias[col0 + 48];

    // C/D layout: lanes 0-15 hold M=r (VGPR r), lanes 16-31 hold M=8+r.
    #pragma unroll
    for (int r = 0; r < 8; ++r) {
        const int g = (mTile << 4) + ((lane >> 4) << 3) + r;    // token row
        const int b_idx = g >> 12;
        const int n     = g & 4095;
        const size_t tokBase = ((size_t)b_idx * NHEAD) * SEQN * HDIM + (size_t)n * HDIM;
        {
            const int c = col0;
            OUT[tokBase + (size_t)(c >> 5) * SEQN * HDIM + (c & 31)] = acc0[r] + bv0;
        }
        {
            const int c = col0 + 16;
            OUT[tokBase + (size_t)(c >> 5) * SEQN * HDIM + (c & 31)] = acc1[r] + bv1;
        }
        {
            const int c = col0 + 32;
            OUT[tokBase + (size_t)(c >> 5) * SEQN * HDIM + (c & 31)] = acc2[r] + bv2;
        }
        {
            const int c = col0 + 48;
            OUT[tokBase + (size_t)(c >> 5) * SEQN * HDIM + (c & 31)] = acc3[r] + bv3;
        }
    }
}

// ---------------------------------------------------------------------------
// Kernel 2: 32-point rfft along head_dim per row; table-driven DFT.
// Output layout [plane][f][n] so the seq FFT and row gathers are coalesced.
__global__ __launch_bounds__(256) void rfft32_rows(
    const float* __restrict__ ql, const float* __restrict__ kl,
    float2* __restrict__ Qf, float2* __restrict__ Kf)
{
    __shared__ float c32[32], s32[32];
    if (threadIdx.x < 32) {
        float s, c;
        __sincosf(TWO_PI * (float)threadIdx.x / 32.0f, &s, &c);
        c32[threadIdx.x] = c; s32[threadIdx.x] = s;
    }
    __syncthreads();

    const int tid = blockIdx.x * 256 + threadIdx.x;      // 0..524287
    const int half = NPLANE * SEQN;                      // 262144
    const float* src = (tid < half) ? ql : kl;
    float2* dst      = (tid < half) ? Qf : Kf;
    const int r = tid & (half - 1);                      // plane*4096 + n
    const int plane = r >> 12;
    const int n     = r & 4095;

    float x[HDIM];
    const float4* p = (const float4*)(src + (size_t)r * HDIM);
    #pragma unroll
    for (int i = 0; i < 8; ++i) {
        float4 f = p[i];
        x[4*i] = f.x; x[4*i+1] = f.y; x[4*i+2] = f.z; x[4*i+3] = f.w;
    }
    for (int f = 0; f < NFREQ; ++f) {
        float re = 0.f, im = 0.f;
        #pragma unroll
        for (int d = 0; d < HDIM; ++d) {
            const int t = (f * d) & 31;
            re += x[d] * c32[t];
            im -= x[d] * s32[t];
        }
        dst[((size_t)plane * NFREQ + f) * SEQN + n] = make_float2(re, im);
    }
}

// ---------------------------------------------------------------------------
// Kernel 3: 4096-point DIF FFT, in LDS, natural -> bit-reversed order.
// (Order irrelevant for convolution: fwd DIF + pointwise mul + inv DIT.)
__global__ __launch_bounds__(256) void fft4096_fwd(float2* __restrict__ Qf,
                                                   float2* __restrict__ Kf)
{
    __shared__ float2 buf[SEQN];                          // 32 KB
    const int blk  = blockIdx.x;                          // 0..2175
    float2* base   = (blk < NPLANE * NFREQ) ? Qf : Kf;
    const int colv = blk % (NPLANE * NFREQ);
    float2* col    = base + (size_t)colv * SEQN;

    for (int i = threadIdx.x; i < SEQN; i += 256) buf[i] = col[i];
    __syncthreads();

    for (int m = SEQN / 2; m >= 1; m >>= 1) {
        for (int t = threadIdx.x; t < SEQN / 2; t += 256) {
            const int j    = t & (m - 1);
            const int blkb = t & ~(m - 1);
            const int i0   = (blkb << 1) + j;
            const int i1   = i0 + m;
            float2 a = buf[i0], b = buf[i1];
            float s, c;
            __sincosf(-PI_F * (float)j / (float)m, &s, &c);
            float2 d = make_float2(a.x - b.x, a.y - b.y);
            buf[i0] = make_float2(a.x + b.x, a.y + b.y);
            buf[i1] = make_float2(d.x * c - d.y * s, d.x * s + d.y * c);
        }
        __syncthreads();
    }
    for (int i = threadIdx.x; i < SEQN; i += 256) col[i] = buf[i];
}

// ---------------------------------------------------------------------------
// Kernel 4: pointwise Qf*Kf, then 4096-point DIT inverse FFT (bit-reversed
// input -> natural output), 1/4096 scaling folded into the store.
__global__ __launch_bounds__(256) void ifft4096_mul(float2* __restrict__ Qf,
                                                    const float2* __restrict__ Kf)
{
    __shared__ float2 buf[SEQN];
    const int colv = blockIdx.x;                           // 0..1087
    float2* qc       = Qf + (size_t)colv * SEQN;
    const float2* kc = Kf + (size_t)colv * SEQN;

    for (int i = threadIdx.x; i < SEQN; i += 256) {
        float2 a = qc[i], b = kc[i];
        buf[i] = make_float2(a.x * b.x - a.y * b.y, a.x * b.y + a.y * b.x);
    }
    __syncthreads();

    for (int m = 1; m <= SEQN / 2; m <<= 1) {
        for (int t = threadIdx.x; t < SEQN / 2; t += 256) {
            const int j    = t & (m - 1);
            const int blkb = t & ~(m - 1);
            const int i0   = (blkb << 1) + j;
            const int i1   = i0 + m;
            float s, c;
            __sincosf(PI_F * (float)j / (float)m, &s, &c);
            float2 a  = buf[i0], b = buf[i1];
            float2 bw = make_float2(b.x * c - b.y * s, b.x * s + b.y * c);
            buf[i0] = make_float2(a.x + bw.x, a.y + bw.y);
            buf[i1] = make_float2(a.x - bw.x, a.y - bw.y);
        }
        __syncthreads();
    }
    const float inv = 1.0f / (float)SEQN;
    for (int i = threadIdx.x; i < SEQN; i += 256) {
        float2 vv = buf[i];
        qc[i] = make_float2(vv.x * inv, vv.y * inv);
    }
}

// ---------------------------------------------------------------------------
// Kernel 5: per-row 32-point inverse rfft (1/32), softmax over 32, gate vl,
// scatter to [B,N,256] interleaved-head layout.
__global__ __launch_bounds__(256) void attn_gate(const float2* __restrict__ Pf,
                                                 const float* __restrict__ vl,
                                                 float* __restrict__ gated)
{
    __shared__ float c32[32], s32[32];
    if (threadIdx.x < 32) {
        float s, c;
        __sincosf(TWO_PI * (float)threadIdx.x / 32.0f, &s, &c);
        c32[threadIdx.x] = c; s32[threadIdx.x] = s;
    }
    __syncthreads();

    const int tid   = blockIdx.x * 256 + threadIdx.x;     // 0..262143
    const int plane = tid >> 12;
    const int n     = tid & 4095;

    float2 P[NFREQ];
    #pragma unroll
    for (int f = 0; f < NFREQ; ++f)
        P[f] = Pf[((size_t)plane * NFREQ + f) * SEQN + n];

    float attn[HDIM];
    const float inv32 = 1.0f / 32.0f;
    for (int d = 0; d < HDIM; ++d) {
        float acc = P[0].x + ((d & 1) ? -P[16].x : P[16].x);
        #pragma unroll
        for (int f = 1; f < 16; ++f) {
            const int t = (f * d) & 31;
            acc += 2.0f * (P[f].x * c32[t] - P[f].y * s32[t]);
        }
        attn[d] = acc * inv32;
    }

    float mx = attn[0];
    #pragma unroll
    for (int d = 1; d < HDIM; ++d) mx = fmaxf(mx, attn[d]);
    float sum = 0.f;
    #pragma unroll
    for (int d = 0; d < HDIM; ++d) { attn[d] = __expf(attn[d] - mx); sum += attn[d]; }
    const float rs = 1.0f / sum;

    const int b_idx = plane >> 3;
    const int h     = plane & 7;
    const float* vrow = vl + (size_t)tid * HDIM;          // [plane][n][32]
    float* orow = gated + ((size_t)b_idx * SEQN + n) * DIMC + h * HDIM;
    #pragma unroll
    for (int d = 0; d < HDIM; ++d) orow[d] = vrow[d] * attn[d] * rs;
}

// ---------------------------------------------------------------------------
// Kernel 6: residual + LayerNorm(256), one wave32 per token.
__global__ __launch_bounds__(256) void residual_ln(const float* __restrict__ gated,
                                                   const float* __restrict__ v,
                                                   const float* __restrict__ gamma,
                                                   const float* __restrict__ beta,
                                                   float* __restrict__ out)
{
    const int lane = threadIdx.x & 31;
    const int tok  = (blockIdx.x * 256 + threadIdx.x) >> 5;   // 0..32767
    const float* g  = gated + (size_t)tok * DIMC;
    const float* vr = v     + (size_t)tok * DIMC;

    float x[8];
    float sum = 0.f, sq = 0.f;
    #pragma unroll
    for (int i = 0; i < 8; ++i) {
        const int c = lane * 8 + i;
        const float val = g[c] + vr[c];
        x[i] = val; sum += val; sq += val * val;
    }
    #pragma unroll
    for (int o = 16; o > 0; o >>= 1) {
        sum += __shfl_xor(sum, o, 32);
        sq  += __shfl_xor(sq,  o, 32);
    }
    const float mu  = sum * (1.0f / 256.0f);
    const float var = sq * (1.0f / 256.0f) - mu * mu;
    const float inv = rsqrtf(var + 1e-5f);

    float* orow = out + (size_t)tok * DIMC;
    #pragma unroll
    for (int i = 0; i < 8; ++i) {
        const int c = lane * 8 + i;
        orow[c] = (x[i] - mu) * inv * gamma[c] + beta[c];
    }
}

// ---------------------------------------------------------------------------
extern "C" void kernel_launch(void* const* d_in, const int* in_sizes, int n_in,
                              void* d_out, int out_size, void* d_ws, size_t ws_size,
                              hipStream_t stream) {
    const float* q     = (const float*)d_in[0];
    const float* k     = (const float*)d_in[1];
    const float* v     = (const float*)d_in[2];
    const float* Wq    = (const float*)d_in[3];
    const float* bq    = (const float*)d_in[4];
    const float* Wk    = (const float*)d_in[5];
    const float* bk    = (const float*)d_in[6];
    const float* Wv    = (const float*)d_in[7];
    const float* bv    = (const float*)d_in[8];
    const float* gamma = (const float*)d_in[9];
    const float* beta  = (const float*)d_in[10];
    float* out = (float*)d_out;

    // Workspace layout (~172 MB):
    const size_t planeElems = (size_t)NPLANE * SEQN * HDIM;     // 8,388,608 floats
    const size_t specElems  = (size_t)NPLANE * NFREQ * SEQN;    // 4,456,448 float2
    char* ws = (char*)d_ws;
    float* ql = (float*)ws;  ws += planeElems * sizeof(float);
    float* kl = (float*)ws;  ws += planeElems * sizeof(float);
    float* vl = (float*)ws;  ws += planeElems * sizeof(float);
    float2* Qf = (float2*)ws; ws += specElems * sizeof(float2);
    float2* Kf = (float2*)ws; ws += specElems * sizeof(float2);
    float* gated = ql;   // ql dead after rfft32_rows; reuse for [B,N,256] gate output

    proj_wmma  <<<dim3(1024, 3), 256, 0, stream>>>(q, k, v, Wq, bq, Wk, bk, Wv, bv,
                                                   ql, kl, vl);
    rfft32_rows<<<2048, 256, 0, stream>>>(ql, kl, Qf, Kf);
    fft4096_fwd<<<2 * NPLANE * NFREQ, 256, 0, stream>>>(Qf, Kf);
    ifft4096_mul<<<NPLANE * NFREQ, 256, 0, stream>>>(Qf, Kf);
    attn_gate  <<<1024, 256, 0, stream>>>(Qf, vl, gated);
    residual_ln<<<4096, 256, 0, stream>>>(gated, v, gamma, beta, out);
}
